// AttentionHead_9380208574842
// MI455X (gfx1250) — matrix operand order repaired and checked by reading, not compiled, not run on hardware
//
#include <hip/hip_runtime.h>

#define HEAD 64
#define EMB  1024
#define BATCH 4
#define SEQ  4096

typedef __attribute__((ext_vector_type(16))) _Float16 v16h;
typedef __attribute__((ext_vector_type(8)))  float    v8f;
typedef __attribute__((ext_vector_type(4)))  float    v4f;

static __device__ inline v8f wmma_f16(v16h a, v16h b, v8f c) {
  return __builtin_amdgcn_wmma_f32_16x16x32_f16(
      /*neg_a=*/false, a, /*neg_b=*/false, b,
      /*c_mod=*/(short)0, c, /*reuse_a=*/false, /*reuse_b=*/false);
}

// A-operand (16x32 f16) from fp32 row-major source, streamed non-temporally.
// lane<16: row=lane, K = k0+{0..15}; lane>=16: K = k0+{16..31}
static __device__ inline v16h load_a_f32_nt(const float* __restrict__ src, int ld,
                                            int k0, int lane) {
  int row = lane & 15;
  int hi  = (lane >> 4) * 16;
  const float* p = src + (size_t)row * ld + k0 + hi;
  v4f x0 = __builtin_nontemporal_load((const v4f*)(p + 0));
  v4f x1 = __builtin_nontemporal_load((const v4f*)(p + 4));
  v4f x2 = __builtin_nontemporal_load((const v4f*)(p + 8));
  v4f x3 = __builtin_nontemporal_load((const v4f*)(p + 12));
  v16h a;
#pragma unroll
  for (int i = 0; i < 4; ++i) {
    a[i]      = (_Float16)x0[i];
    a[4 + i]  = (_Float16)x1[i];
    a[8 + i]  = (_Float16)x2[i];
    a[12 + i] = (_Float16)x3[i];
  }
  return a;
}

// A-operand (16x32 f16) from f16 row-major source.
static __device__ inline v16h load_a_f16(const _Float16* __restrict__ src, int ld,
                                         int k0, int lane) {
  int row = lane & 15;
  int hi  = (lane >> 4) * 16;
  const _Float16* p = src + (size_t)row * ld + k0 + hi;
  v16h a;
#pragma unroll
  for (int i = 0; i < 8; ++i) a[i] = p[i];
#pragma unroll
  for (int i = 0; i < 8; ++i) a[8 + i] = p[8 + i];
  return a;
}

// B-operand (32x16) from pre-transposed f16 weights WT[HEAD x EMB].
// lane<16: col = n0+lane (row of WT), K rows k0+0..15 (contiguous);
// lane>=16: same col, K rows k0+16..31. Single 32-byte vector load.
static __device__ inline v16h load_b_WT(const _Float16* __restrict__ WT, int k0,
                                        int n0, int lane) {
  int col = n0 + (lane & 15);
  int kk  = k0 + (lane >> 4) * 16;
  return *(const v16h*)(WT + (size_t)col * EMB + kk);
}

// B-operand for Q*K^T: B = K^T[64 x 16]; source k_h row-major [SEQ x HEAD].
static __device__ inline v16h load_b_kT(const _Float16* __restrict__ kb,
                                        int tok0, int d0, int lane) {
  int tok = tok0 + (lane & 15);
  int d   = d0 + (lane >> 4) * 16;
  return *(const v16h*)(kb + (size_t)tok * HEAD + d);
}

// B-operand for P*V: B = V[32 keys x 16 dims]; source vT row-major [HEAD x SEQ].
static __device__ inline v16h load_b_vT(const _Float16* __restrict__ vb,
                                        int n0, int k0, int lane) {
  int n  = n0 + (lane & 15);
  int kk = k0 + (lane >> 4) * 16;
  return *(const v16h*)(vb + (size_t)n * SEQ + kk);
}

// ---------------- Phase 0: pack weights fp32 [EMB x HEAD] -> f16 WT [HEAD x EMB]
__global__ __launch_bounds__(256) void w_pack_kernel(
    const float* __restrict__ Wq, const float* __restrict__ Wk,
    const float* __restrict__ Wv, _Float16* __restrict__ WqT,
    _Float16* __restrict__ WkT, _Float16* __restrict__ WvT) {
  int z = blockIdx.y;
  const float* W = (z == 0) ? Wq : (z == 1) ? Wk : Wv;
  _Float16* WT   = (z == 0) ? WqT : (z == 1) ? WkT : WvT;
  int idx = blockIdx.x * 256 + threadIdx.x;   // 0 .. HEAD*EMB-1
  int h = idx >> 10;                          // / EMB
  int e = idx & (EMB - 1);
  WT[(size_t)h * EMB + e] = (_Float16)W[(size_t)e * HEAD + h];
}

// ---------------- Phase 1: fully fused QKV projection.
// One wave owns one 16-row block of x and produces ALL 12 output tiles
// (3 matrices x 4 n-tiles): each x element is loaded exactly once from HBM.
__global__ __launch_bounds__(256) void qkv_proj_kernel(
    const float* __restrict__ x, const _Float16* __restrict__ WqT,
    const _Float16* __restrict__ WkT, const _Float16* __restrict__ WvT,
    _Float16* __restrict__ qh, _Float16* __restrict__ kh,
    _Float16* __restrict__ vT) {
  int lane = threadIdx.x & 31;
  int wv   = threadIdx.x >> 5;
  int mt   = blockIdx.x * 8 + wv;   // 0..(B*T/16 - 1)
  int row0 = mt * 16;

  const _Float16* WTs[3] = {WqT, WkT, WvT};
  const float* xa = x + (size_t)row0 * EMB;

  v8f acc[12] = {};                 // [z*4 + nt]
#pragma unroll 1
  for (int k0 = 0; k0 < EMB; k0 += 32) {
    v16h a = load_a_f32_nt(xa, EMB, k0, lane);
#pragma unroll
    for (int z = 0; z < 3; ++z) {
#pragma unroll
      for (int nt = 0; nt < 4; ++nt) {
        v16h b = load_b_WT(WTs[z], k0, nt * 16, lane);
        acc[z * 4 + nt] = wmma_f16(a, b, acc[z * 4 + nt]);
      }
    }
  }

  int mloc = (lane >> 4) * 8;
  int ncol = lane & 15;
  // q and k: row-major f16 [B*SEQ x HEAD]
#pragma unroll
  for (int z = 0; z < 2; ++z) {
    _Float16* dst = (z == 0) ? qh : kh;
#pragma unroll
    for (int nt = 0; nt < 4; ++nt)
#pragma unroll
      for (int r = 0; r < 8; ++r)
        dst[(size_t)(row0 + mloc + r) * HEAD + nt * 16 + ncol] =
            (_Float16)acc[z * 4 + nt][r];
  }
  // v: transposed f16 [B][HEAD x SEQ]
  int bb = row0 >> 12;              // SEQ = 4096
  int t0 = row0 & (SEQ - 1);
#pragma unroll
  for (int nt = 0; nt < 4; ++nt)
#pragma unroll
    for (int r = 0; r < 8; ++r)
      vT[(size_t)(bb * HEAD + nt * 16 + ncol) * SEQ + t0 + mloc + r] =
          (_Float16)acc[8 + nt][r];
}

// ---------------- Phase 2: causal flash attention, one wave per 16-query tile
__global__ __launch_bounds__(256) void flash_attn_kernel(
    const _Float16* __restrict__ qh, const _Float16* __restrict__ kh,
    const _Float16* __restrict__ vT, float* __restrict__ out) {
  __shared__ _Float16 Plds[8][16][32];   // per-wave P bounce buffer (C->A layout)

  int lane = threadIdx.x & 31;
  int wv   = threadIdx.x >> 5;
  int qt   = blockIdx.x * 8 + wv;        // 0..SEQ/16-1
  int b    = blockIdx.y;
  int q0   = qt * 16;

  const _Float16* qbp = qh + ((size_t)b * SEQ + q0) * HEAD;
  const _Float16* kbp = kh + (size_t)b * SEQ * HEAD;
  const _Float16* vbp = vT + (size_t)b * HEAD * SEQ;

  v16h qa0 = load_a_f16(qbp, HEAD, 0, lane);
  v16h qa1 = load_a_f16(qbp, HEAD, 32, lane);

  v8f acc0 = {}, acc1 = {}, acc2 = {}, acc3 = {};
  float mrow[8], lrow[8];
#pragma unroll
  for (int r = 0; r < 8; ++r) { mrow[r] = -1e30f; lrow[r] = 0.0f; }

  int mloc = (lane >> 4) * 8;
  int ncol = lane & 15;

  for (int k0 = 0; k0 <= q0 + 15; k0 += 32) {
    // prefetch next key block of K and V (speculative; L2 -> WGP overlap)
    __builtin_prefetch(kbp + (size_t)(k0 + 32) * HEAD + lane * 4, 0, 1);
    __builtin_prefetch(vbp + (size_t)(lane * 2) * SEQ + k0 + 32, 0, 1);

    // ---- scores S[16q x 32k] = Q K^T (two 16x16 C tiles, K-dim 64 = 2 steps)
    v8f s0 = {}, s1 = {};
    s0 = wmma_f16(qa0, load_b_kT(kbp, k0,       0, lane), s0);
    s0 = wmma_f16(qa1, load_b_kT(kbp, k0,      32, lane), s0);
    s1 = wmma_f16(qa0, load_b_kT(kbp, k0 + 16,  0, lane), s1);
    s1 = wmma_f16(qa1, load_b_kT(kbp, k0 + 16, 32, lane), s1);

    // ---- scale 1/sqrt(64) + causal mask
#pragma unroll
    for (int r = 0; r < 8; ++r) {
      int row = q0 + mloc + r;
      float v0 = s0[r] * 0.125f;
      float v1 = s1[r] * 0.125f;
      if (k0 + ncol > row)      v0 = -1e30f;
      if (k0 + 16 + ncol > row) v1 = -1e30f;
      s0[r] = v0; s1[r] = v1;
    }

    // ---- online softmax (row stats live across 16-lane halves)
    float scale[8];
#pragma unroll
    for (int r = 0; r < 8; ++r) {
      float mx = fmaxf(s0[r], s1[r]);
#pragma unroll
      for (int off = 1; off < 16; off <<= 1)
        mx = fmaxf(mx, __shfl_xor(mx, off, 32));
      float mn = fmaxf(mrow[r], mx);
      scale[r] = __expf(mrow[r] - mn);
      float p0 = __expf(s0[r] - mn);
      float p1 = __expf(s1[r] - mn);
      s0[r] = p0; s1[r] = p1;
      float sum = p0 + p1;
#pragma unroll
      for (int off = 1; off < 16; off <<= 1)
        sum += __shfl_xor(sum, off, 32);
      lrow[r] = lrow[r] * scale[r] + sum;
      mrow[r] = mn;
    }

    // ---- rescale running output
#pragma unroll
    for (int r = 0; r < 8; ++r) {
      acc0[r] *= scale[r]; acc1[r] *= scale[r];
      acc2[r] *= scale[r]; acc3[r] *= scale[r];
    }

    // ---- P: C-layout -> A-layout via LDS (per-wave, DS ops are in-order)
#pragma unroll
    for (int r = 0; r < 8; ++r) {
      Plds[wv][mloc + r][ncol]      = (_Float16)s0[r];
      Plds[wv][mloc + r][ncol + 16] = (_Float16)s1[r];
    }
    asm volatile("s_wait_dscnt 0" ::: "memory");
    v16h pa;
    {
      const _Float16* p = &Plds[wv][lane & 15][(lane >> 4) * 16];
#pragma unroll
      for (int i = 0; i < 8; ++i) pa[i] = p[i];
#pragma unroll
      for (int i = 0; i < 8; ++i) pa[8 + i] = p[8 + i];
    }

    // ---- O += P V   (V as 4 B-tiles of 16 head dims, contiguous from vT)
    acc0 = wmma_f16(pa, load_b_vT(vbp,  0, k0, lane), acc0);
    acc1 = wmma_f16(pa, load_b_vT(vbp, 16, k0, lane), acc1);
    acc2 = wmma_f16(pa, load_b_vT(vbp, 32, k0, lane), acc2);
    acc3 = wmma_f16(pa, load_b_vT(vbp, 48, k0, lane), acc3);
  }

  // ---- finalize: out = O / l   (fp32 output [B, T, HEAD])
  float* ob = out + ((size_t)b * SEQ + q0) * HEAD;
#pragma unroll
  for (int r = 0; r < 8; ++r) {
    float inv = 1.0f / lrow[r];
    size_t m = (size_t)(mloc + r) * HEAD;
    ob[m + 0  + ncol] = acc0[r] * inv;
    ob[m + 16 + ncol] = acc1[r] * inv;
    ob[m + 32 + ncol] = acc2[r] * inv;
    ob[m + 48 + ncol] = acc3[r] * inv;
  }
}

extern "C" void kernel_launch(void* const* d_in, const int* in_sizes, int n_in,
                              void* d_out, int out_size, void* d_ws, size_t ws_size,
                              hipStream_t stream) {
  const float* x  = (const float*)d_in[0];
  const float* Wq = (const float*)d_in[1];
  const float* Wk = (const float*)d_in[2];
  const float* Wv = (const float*)d_in[3];
  float* out = (float*)d_out;

  char* ws = (char*)d_ws;
  const size_t QKV_BYTES = (size_t)BATCH * SEQ * HEAD * sizeof(_Float16); // 2 MiB
  const size_t WT_BYTES  = (size_t)HEAD * EMB * sizeof(_Float16);        // 128 KiB
  _Float16* qh  = (_Float16*)(ws);
  _Float16* kh  = (_Float16*)(ws + QKV_BYTES);
  _Float16* vT  = (_Float16*)(ws + 2 * QKV_BYTES);
  _Float16* WqT = (_Float16*)(ws + 3 * QKV_BYTES);
  _Float16* WkT = (_Float16*)(ws + 3 * QKV_BYTES + WT_BYTES);
  _Float16* WvT = (_Float16*)(ws + 3 * QKV_BYTES + 2 * WT_BYTES);

  dim3 g0(HEAD * EMB / 256, 3, 1);
  w_pack_kernel<<<g0, 256, 0, stream>>>(Wq, Wk, Wv, WqT, WkT, WvT);

  dim3 g1(BATCH * SEQ / 16 / 8, 1, 1);   // 128 blocks x 8 waves: 1 wave per 16 rows
  qkv_proj_kernel<<<g1, 256, 0, stream>>>(x, WqT, WkT, WvT, qh, kh, vT);

  dim3 g2(SEQ / 16 / 8, BATCH, 1);       // 32 x 4 blocks, 8 waves
  flash_attn_kernel<<<g2, 256, 0, stream>>>(qh, kh, vT, out);
}
